// attention_51908974739903
// MI455X (gfx1250) — matrix operand order repaired and checked by reading, not compiled, not run on hardware
//
#include <hip/hip_runtime.h>

typedef __bf16 bf16_t;
typedef __attribute__((ext_vector_type(16))) __bf16 v16bf;
typedef __attribute__((ext_vector_type(8)))  __bf16 v8bf;
typedef __attribute__((ext_vector_type(8)))  float   v8f;

#define EN 256
#define HN 8
#define DN 32
#define BN 8
#define SN 1024
#define TN 1025
#define TPAD 1056                   // T padded to multiple of 32
#define MPAD (BN*TPAD)              // 8448 rows
#define NBIG ((size_t)MPAD*EN)      // 2,162,688 elems

#define V8Z {0.f,0.f,0.f,0.f,0.f,0.f,0.f,0.f}

union Frag { v16bf v; v8bf h[2]; };

__device__ __forceinline__ v8bf ld8(const bf16_t* p) { return *(const v8bf*)p; }

__device__ __forceinline__ v8f wmma_bf(Frag a, Frag b, v8f c) {
  return __builtin_amdgcn_wmma_f32_16x16x32_bf16(false, a.v, false, b.v, (short)0, c, false, false);
}

// ---------------- pack X = concat(q, y) along seq, padded, bf16 (+ negated imag)
__global__ void pack_x(const float* __restrict__ qr, const float* __restrict__ qi,
                       const float* __restrict__ yr, const float* __restrict__ yi,
                       bf16_t* __restrict__ XR, bf16_t* __restrict__ XI, bf16_t* __restrict__ XIN) {
  size_t idx = (size_t)blockIdx.x * 256 + threadIdx.x;     // over NBIG
  int e = (int)(idx & (EN - 1));
  int gm = (int)(idx >> 8);
  int b = gm / TPAD, t = gm % TPAD;
  float vr = 0.f, vi = 0.f;
  if (t < SN)       { size_t s = ((size_t)b * SN + t) * EN + e; vr = qr[s]; vi = qi[s]; }
  else if (t == SN) { size_t s = (size_t)b * EN + e;            vr = yr[s]; vi = yi[s]; }
  XR[idx]  = (bf16_t)vr;
  XI[idx]  = (bf16_t)vi;
  XIN[idx] = (bf16_t)(-vi);
}

// ---------------- convert one weight pair to bf16 (row-major [n][k] == W^T as B operand)
__global__ void conv_w(const float* __restrict__ wr, const float* __restrict__ wi,
                       bf16_t* __restrict__ WR, bf16_t* __restrict__ WI) {
  int idx = blockIdx.x * 256 + threadIdx.x;                // 65536 elems
  WR[idx] = (bf16_t)wr[idx];
  WI[idx] = (bf16_t)wi[idx];
}

// ---------------- fused complex projection GEMM: Z = X W^T + b for Q/K/V (blockIdx.y selects)
// block = 8 waves; wave w owns N columns [w*32, w*32+32) (== head w), block owns 16 M rows
__global__ __launch_bounds__(256) void proj_qkv(
    const bf16_t* __restrict__ XR, const bf16_t* __restrict__ XI, const bf16_t* __restrict__ XIN,
    const bf16_t* __restrict__ WqR, const bf16_t* __restrict__ WqI,
    const bf16_t* __restrict__ WkR, const bf16_t* __restrict__ WkI,
    const bf16_t* __restrict__ WvR, const bf16_t* __restrict__ WvI,
    const float* __restrict__ bqr, const float* __restrict__ bqi,
    const float* __restrict__ bkr, const float* __restrict__ bki,
    const float* __restrict__ bvr, const float* __restrict__ bvi,
    bf16_t* __restrict__ QR, bf16_t* __restrict__ QI,
    bf16_t* __restrict__ KR, bf16_t* __restrict__ KI, bf16_t* __restrict__ KIN,
    bf16_t* __restrict__ VRT, bf16_t* __restrict__ VIT) {
  int z = blockIdx.y;
  const bf16_t* WR = (z == 0) ? WqR : (z == 1) ? WkR : WvR;
  const bf16_t* WI = (z == 0) ? WqI : (z == 1) ? WkI : WvI;
  const float*  br = (z == 0) ? bqr : (z == 1) ? bkr : bvr;
  const float*  bi = (z == 0) ? bqi : (z == 1) ? bki : bvi;

  int wave = threadIdx.x >> 5, lane = threadIdx.x & 31;
  int half = lane >> 4, lm = lane & 15;
  int m0 = blockIdx.x * 16;

  v8f accR0 = V8Z, accR1 = V8Z, accI0 = V8Z, accI1 = V8Z;
  const size_t arow = (size_t)(m0 + lm) * EN;
  const size_t brow0 = (size_t)(wave * 32 + lm) * EN;
  const size_t brow1 = (size_t)(wave * 32 + 16 + lm) * EN;

#pragma unroll
  for (int kt = 0; kt < 8; ++kt) {
    int ka = kt * 32 + half * 8;       // A-frag: elems 0-7 at ka, 8-15 at ka+16
    int kb = kt * 32 + half * 8;       // B-frag: elems 0-7 at kb, 8-15 at kb+16
    Frag Ar, Ai, An, Br0, Bi0, Br1, Bi1;
    Ar.h[0] = ld8(XR + arow + ka);  Ar.h[1] = ld8(XR + arow + ka + 16);
    Ai.h[0] = ld8(XI + arow + ka);  Ai.h[1] = ld8(XI + arow + ka + 16);
    An.h[0] = ld8(XIN + arow + ka); An.h[1] = ld8(XIN + arow + ka + 16);
    Br0.h[0] = ld8(WR + brow0 + kb); Br0.h[1] = ld8(WR + brow0 + kb + 16);
    Bi0.h[0] = ld8(WI + brow0 + kb); Bi0.h[1] = ld8(WI + brow0 + kb + 16);
    Br1.h[0] = ld8(WR + brow1 + kb); Br1.h[1] = ld8(WR + brow1 + kb + 16);
    Bi1.h[0] = ld8(WI + brow1 + kb); Bi1.h[1] = ld8(WI + brow1 + kb + 16);
    accR0 = wmma_bf(Ar, Br0, accR0); accR0 = wmma_bf(An, Bi0, accR0);   // Xr*Wr - Xi*Wi
    accI0 = wmma_bf(Ar, Bi0, accI0); accI0 = wmma_bf(Ai, Br0, accI0);   // Xr*Wi + Xi*Wr
    accR1 = wmma_bf(Ar, Br1, accR1); accR1 = wmma_bf(An, Bi1, accR1);
    accI1 = wmma_bf(Ar, Bi1, accI1); accI1 = wmma_bf(Ai, Br1, accI1);
  }

  int h = wave;
  float brv0 = br[wave * 32 + lm],      biv0 = bi[wave * 32 + lm];
  float brv1 = br[wave * 32 + 16 + lm], biv1 = bi[wave * 32 + 16 + lm];

#pragma unroll
  for (int r = 0; r < 8; ++r) {
    int gm = m0 + r + 8 * half;
    int b_ = gm / TPAD, t = gm % TPAD;
    float zr0 = accR0[r] + brv0, zi0 = accI0[r] + biv0;
    float zr1 = accR1[r] + brv1, zi1 = accI1[r] + biv1;
    if (z == 0) {
      size_t o = ((size_t)(b_ * HN + h) * TPAD + t) * DN;
      QR[o + lm] = (bf16_t)zr0;      QI[o + lm] = (bf16_t)zi0;
      QR[o + 16 + lm] = (bf16_t)zr1; QI[o + 16 + lm] = (bf16_t)zi1;
    } else if (z == 1) {
      size_t o = ((size_t)(b_ * HN + h) * TPAD + t) * DN;
      KR[o + lm] = (bf16_t)zr0;      KI[o + lm] = (bf16_t)zi0;      KIN[o + lm] = (bf16_t)(-zi0);
      KR[o + 16 + lm] = (bf16_t)zr1; KI[o + 16 + lm] = (bf16_t)zi1; KIN[o + 16 + lm] = (bf16_t)(-zi1);
    } else {                                  // V transposed: [b,h,d,t]
      size_t o = (size_t)(b_ * HN + h) * DN;
      VRT[(o + lm) * TPAD + t] = (bf16_t)zr0;      VIT[(o + lm) * TPAD + t] = (bf16_t)zi0;
      VRT[(o + 16 + lm) * TPAD + t] = (bf16_t)zr1; VIT[(o + 16 + lm) * TPAD + t] = (bf16_t)zi1;
    }
  }
}

// ---------------- flash attention: one wave per 16-query tile of one (h,b)
__global__ __launch_bounds__(256) void attn_flash(
    const bf16_t* __restrict__ QR, const bf16_t* __restrict__ QI,
    const bf16_t* __restrict__ KR, const bf16_t* __restrict__ KI, const bf16_t* __restrict__ KIN,
    const bf16_t* __restrict__ VRT, const bf16_t* __restrict__ VIT,
    bf16_t* __restrict__ HR, bf16_t* __restrict__ HI, bf16_t* __restrict__ HIN) {
  __shared__ __align__(16) bf16_t sP[8][16 * 32];

  int wave = threadIdx.x >> 5, lane = threadIdx.x & 31;
  int half = lane >> 4, lm = lane & 15;
  int g = blockIdx.x * 8 + wave;
  int qt = g % (TPAD / 16);            // 66 query tiles
  int hb = g / (TPAD / 16);
  int h = hb & 7, b = hb >> 3;

  const size_t base  = (size_t)(b * HN + h) * TPAD * DN;   // Q,K row-major [t][d]
  const size_t vbase = (size_t)(b * HN + h) * DN * TPAD;   // V transposed [d][t]
  const bf16_t* Qr = QR + base;  const bf16_t* Qi = QI + base;
  const bf16_t* Kr = KR + base;  const bf16_t* Ki = KI + base;  const bf16_t* Kn = KIN + base;
  const bf16_t* Vr = VRT + vbase; const bf16_t* Vi = VIT + vbase;

  Frag Qrf, Qif;
  {
    size_t q0 = (size_t)(qt * 16 + lm) * DN + half * 8;
    Qrf.h[0] = ld8(Qr + q0); Qrf.h[1] = ld8(Qr + q0 + 16);
    Qif.h[0] = ld8(Qi + q0); Qif.h[1] = ld8(Qi + q0 + 16);
  }

  float rmax[8], rsum[8];
#pragma unroll
  for (int r = 0; r < 8; ++r) { rmax[r] = -3e38f; rsum[r] = 0.f; }
  v8f aR0 = V8Z, aR1 = V8Z, aI0 = V8Z, aI1 = V8Z;

  for (int kb = 0; kb < TPAD; kb += 32) {
#pragma unroll
    for (int kk = 0; kk < 2; ++kk) {
      int k16 = kb + kk * 16;
      Frag Krf, Kif, Knf;
      size_t kr = (size_t)(k16 + lm) * DN + half * 8;
      Krf.h[0] = ld8(Kr + kr); Krf.h[1] = ld8(Kr + kr + 16);
      Kif.h[0] = ld8(Ki + kr); Kif.h[1] = ld8(Ki + kr + 16);
      Knf.h[0] = ld8(Kn + kr); Knf.h[1] = ld8(Kn + kr + 16);
      v8f zero = V8Z;
      v8f sr = wmma_bf(Qrf, Krf, zero); sr = wmma_bf(Qif, Kif, sr);   // Re(Q conj(K)^T)
      v8f si = wmma_bf(Qif, Krf, zero); si = wmma_bf(Qrf, Knf, si);   // Im(Q conj(K)^T)
      bool valid = (k16 + lm) < TN;
#pragma unroll
      for (int r = 0; r < 8; ++r) {
        float a = sr[r], c = si[r];
        float l = valid ? 10.f * a * rsqrtf(a * a + c * c + 1e-30f) : -1e30f;
        float tm = l;
        tm = fmaxf(tm, __shfl_xor(tm, 1, 32));
        tm = fmaxf(tm, __shfl_xor(tm, 2, 32));
        tm = fmaxf(tm, __shfl_xor(tm, 4, 32));
        tm = fmaxf(tm, __shfl_xor(tm, 8, 32));
        float mn = fmaxf(rmax[r], tm);
        float sc = __expf(rmax[r] - mn);
        float p  = __expf(l - mn);
        float ps = p;
        ps += __shfl_xor(ps, 1, 32); ps += __shfl_xor(ps, 2, 32);
        ps += __shfl_xor(ps, 4, 32); ps += __shfl_xor(ps, 8, 32);
        rsum[r] = rsum[r] * sc + ps;
        rmax[r] = mn;
        aR0[r] *= sc; aR1[r] *= sc; aI0[r] *= sc; aI1[r] *= sc;
        sP[wave][(r + 8 * half) * 32 + kk * 16 + lm] = (bf16_t)p;     // transpose via LDS
      }
    }
    __syncthreads();
    Frag Wf;
    {
      const bf16_t* wrow = &sP[wave][lm * 32 + half * 8];
      Wf.h[0] = *(const v8bf*)wrow; Wf.h[1] = *(const v8bf*)(wrow + 16);
    }
    Frag Vr0, Vr1, Vi0, Vi1;
    size_t v0 = (size_t)lm * TPAD + kb + half * 8;
    size_t v1 = (size_t)(16 + lm) * TPAD + kb + half * 8;
    Vr0.h[0] = ld8(Vr + v0); Vr0.h[1] = ld8(Vr + v0 + 16);
    Vr1.h[0] = ld8(Vr + v1); Vr1.h[1] = ld8(Vr + v1 + 16);
    Vi0.h[0] = ld8(Vi + v0); Vi0.h[1] = ld8(Vi + v0 + 16);
    Vi1.h[0] = ld8(Vi + v1); Vi1.h[1] = ld8(Vi + v1 + 16);
    aR0 = wmma_bf(Wf, Vr0, aR0); aR1 = wmma_bf(Wf, Vr1, aR1);
    aI0 = wmma_bf(Wf, Vi0, aI0); aI1 = wmma_bf(Wf, Vi1, aI1);
    __syncthreads();
  }

#pragma unroll
  for (int r = 0; r < 8; ++r) {
    int t = qt * 16 + r + 8 * half;
    if (t < TN) {
      float inv = 1.f / rsum[r];
      size_t o = ((size_t)b * TPAD + t) * EN + h * DN;
      float r0 = aR0[r] * inv, i0 = aI0[r] * inv;
      float r1 = aR1[r] * inv, i1 = aI1[r] * inv;
      HR[o + lm] = (bf16_t)r0;      HI[o + lm] = (bf16_t)i0;      HIN[o + lm] = (bf16_t)(-i0);
      HR[o + 16 + lm] = (bf16_t)r1; HI[o + 16 + lm] = (bf16_t)i1; HIN[o + 16 + lm] = (bf16_t)(-i1);
    }
  }
}

// ---------------- output complex GEMM: out = heads @ Wo^T + bo, interleaved complex64
__global__ __launch_bounds__(256) void out_proj(
    const bf16_t* __restrict__ HRp, const bf16_t* __restrict__ HIp, const bf16_t* __restrict__ HNp,
    const bf16_t* __restrict__ WR, const bf16_t* __restrict__ WI,
    const float* __restrict__ br, const float* __restrict__ bi,
    float* __restrict__ out) {
  int wave = threadIdx.x >> 5, lane = threadIdx.x & 31;
  int half = lane >> 4, lm = lane & 15;
  int m0 = blockIdx.x * 16;

  v8f accR0 = V8Z, accR1 = V8Z, accI0 = V8Z, accI1 = V8Z;
  const size_t arow = (size_t)(m0 + lm) * EN;
  const size_t brow0 = (size_t)(wave * 32 + lm) * EN;
  const size_t brow1 = (size_t)(wave * 32 + 16 + lm) * EN;

#pragma unroll
  for (int kt = 0; kt < 8; ++kt) {
    int ka = kt * 32 + half * 8;
    int kb = kt * 32 + half * 8;
    Frag Ar, Ai, An, Br0, Bi0, Br1, Bi1;
    Ar.h[0] = ld8(HRp + arow + ka); Ar.h[1] = ld8(HRp + arow + ka + 16);
    Ai.h[0] = ld8(HIp + arow + ka); Ai.h[1] = ld8(HIp + arow + ka + 16);
    An.h[0] = ld8(HNp + arow + ka); An.h[1] = ld8(HNp + arow + ka + 16);
    Br0.h[0] = ld8(WR + brow0 + kb); Br0.h[1] = ld8(WR + brow0 + kb + 16);
    Bi0.h[0] = ld8(WI + brow0 + kb); Bi0.h[1] = ld8(WI + brow0 + kb + 16);
    Br1.h[0] = ld8(WR + brow1 + kb); Br1.h[1] = ld8(WR + brow1 + kb + 16);
    Bi1.h[0] = ld8(WI + brow1 + kb); Bi1.h[1] = ld8(WI + brow1 + kb + 16);
    accR0 = wmma_bf(Ar, Br0, accR0); accR0 = wmma_bf(An, Bi0, accR0);
    accI0 = wmma_bf(Ar, Bi0, accI0); accI0 = wmma_bf(Ai, Br0, accI0);
    accR1 = wmma_bf(Ar, Br1, accR1); accR1 = wmma_bf(An, Bi1, accR1);
    accI1 = wmma_bf(Ar, Bi1, accI1); accI1 = wmma_bf(Ai, Br1, accI1);
  }

  int n0 = wave * 32 + lm;
  float brv0 = br[n0], biv0 = bi[n0], brv1 = br[n0 + 16], biv1 = bi[n0 + 16];
  float2* o2 = (float2*)out;

#pragma unroll
  for (int r = 0; r < 8; ++r) {
    int gm = m0 + r + 8 * half;
    int b_ = gm / TPAD, t = gm % TPAD;
    float2 z0 = { accR0[r] + brv0, accI0[r] + biv0 };
    float2 z1 = { accR1[r] + brv1, accI1[r] + biv1 };
    if (t < SN) {                         // first output: [B,S,E] complex
      size_t o = (size_t)(b_ * SN + t) * EN;
      o2[o + n0] = z0; o2[o + n0 + 16] = z1;
    } else if (t == SN) {                 // second output: [B,1,E] complex
      size_t o = (size_t)BN * SN * EN + (size_t)b_ * EN;
      o2[o + n0] = z0; o2[o + n0 + 16] = z1;
    }
  }
}

extern "C" void kernel_launch(void* const* d_in, const int* in_sizes, int n_in,
                              void* d_out, int out_size, void* d_ws, size_t ws_size,
                              hipStream_t stream) {
  const float* q_r = (const float*)d_in[0];
  const float* q_i = (const float*)d_in[1];
  const float* y_r = (const float*)d_in[2];
  const float* y_i = (const float*)d_in[3];
  const float* wq_r = (const float*)d_in[4];  const float* wq_i = (const float*)d_in[5];
  const float* bq_r = (const float*)d_in[6];  const float* bq_i = (const float*)d_in[7];
  const float* wk_r = (const float*)d_in[8];  const float* wk_i = (const float*)d_in[9];
  const float* bk_r = (const float*)d_in[10]; const float* bk_i = (const float*)d_in[11];
  const float* wv_r = (const float*)d_in[12]; const float* wv_i = (const float*)d_in[13];
  const float* bv_r = (const float*)d_in[14]; const float* bv_i = (const float*)d_in[15];
  const float* wo_r = (const float*)d_in[16]; const float* wo_i = (const float*)d_in[17];
  const float* bo_r = (const float*)d_in[18]; const float* bo_i = (const float*)d_in[19];

  char* ws = (char*)d_ws;
  size_t off = 0;
  auto alloc = [&](size_t bytes) -> void* {
    off = (off + 255) & ~(size_t)255;
    void* p = (void*)(ws + off);
    off += bytes;
    return p;
  };
  const size_t big = NBIG * sizeof(bf16_t);
  const size_t wsz = (size_t)EN * EN * sizeof(bf16_t);
  bf16_t* XR  = (bf16_t*)alloc(big);  bf16_t* XI  = (bf16_t*)alloc(big);  bf16_t* XIN = (bf16_t*)alloc(big);
  bf16_t* WqR = (bf16_t*)alloc(wsz);  bf16_t* WqI = (bf16_t*)alloc(wsz);
  bf16_t* WkR = (bf16_t*)alloc(wsz);  bf16_t* WkI = (bf16_t*)alloc(wsz);
  bf16_t* WvR = (bf16_t*)alloc(wsz);  bf16_t* WvI = (bf16_t*)alloc(wsz);
  bf16_t* WoR = (bf16_t*)alloc(wsz);  bf16_t* WoI = (bf16_t*)alloc(wsz);
  bf16_t* QR  = (bf16_t*)alloc(big);  bf16_t* QI  = (bf16_t*)alloc(big);
  bf16_t* KR  = (bf16_t*)alloc(big);  bf16_t* KI  = (bf16_t*)alloc(big);  bf16_t* KIN = (bf16_t*)alloc(big);
  bf16_t* VRT = (bf16_t*)alloc(big);  bf16_t* VIT = (bf16_t*)alloc(big);
  bf16_t* HR  = (bf16_t*)alloc(big);  bf16_t* HI  = (bf16_t*)alloc(big);  bf16_t* HIN = (bf16_t*)alloc(big);
  (void)ws_size; (void)n_in; (void)in_sizes; (void)out_size;

  pack_x<<<MPAD, 256, 0, stream>>>(q_r, q_i, y_r, y_i, XR, XI, XIN);
  conv_w<<<EN * EN / 256, 256, 0, stream>>>(wq_r, wq_i, WqR, WqI);
  conv_w<<<EN * EN / 256, 256, 0, stream>>>(wk_r, wk_i, WkR, WkI);
  conv_w<<<EN * EN / 256, 256, 0, stream>>>(wv_r, wv_i, WvR, WvI);
  conv_w<<<EN * EN / 256, 256, 0, stream>>>(wo_r, wo_i, WoR, WoI);

  proj_qkv<<<dim3(MPAD / 16, 3), 256, 0, stream>>>(
      XR, XI, XIN, WqR, WqI, WkR, WkI, WvR, WvI,
      bq_r, bq_i, bk_r, bk_i, bv_r, bv_i,
      QR, QI, KR, KI, KIN, VRT, VIT);

  attn_flash<<<(TPAD / 16) * HN * BN / 8, 256, 0, stream>>>(
      QR, QI, KR, KI, KIN, VRT, VIT, HR, HI, HIN);

  out_proj<<<MPAD / 16, 256, 0, stream>>>(
      HR, HI, HIN, WoR, WoI, bo_r, bo_i, (float*)d_out);
}
